// TransformerEncoderLayer_85332410237164
// MI455X (gfx1250) — matrix operand order repaired and checked by reading, not compiled, not run on hardware
//
#include <hip/hip_runtime.h>
#include <cstddef>

// ---------------------------------------------------------------------------
// CDNA5 (gfx1250) transformer encoder layer, bf16 WMMA end-to-end.
//   B=4, S=1024, D=1024, H=16, HD=64, FF=4096. fp32 residual stream,
//   bf16 weights/activations, f32 accumulation on the matrix cores.
//   Tile staging: double-buffered GLOBAL_LOAD_ASYNC_TO_LDS_B128 (ASYNCcnt),
//   overlapping LDS-DMA with WMMA compute.
// ---------------------------------------------------------------------------

#define USE_ASYNC 1

typedef __attribute__((ext_vector_type(16))) __bf16 bf16x16;
typedef __attribute__((ext_vector_type(8)))  float  f32x8;

union BFrag { bf16x16 v; unsigned u[8]; };

static __device__ __forceinline__ unsigned short f2bf(float f) {
  unsigned u = __builtin_bit_cast(unsigned, f);
  u += 0x7FFFu + ((u >> 16) & 1u);            // round-to-nearest-even
  return (unsigned short)(u >> 16);
}
static __device__ __forceinline__ unsigned pack2(float a, float b) {
  return (unsigned)f2bf(a) | ((unsigned)f2bf(b) << 16);
}
static __device__ __forceinline__ f32x8 wmma_bf16(const BFrag& a, const BFrag& b, f32x8 c) {
  return __builtin_amdgcn_wmma_f32_16x16x32_bf16(false, a.v, false, b.v,
                                                 (short)0, c, false, false);
}
static __device__ __forceinline__ float gelu_exact(float x) {
  return 0.5f * x * (1.0f + erff(x * 0.70710678118654752f));
}

// 16-byte global -> LDS copy via the CDNA5 LDS-DMA instruction (ASYNCcnt).
static __device__ __forceinline__ void cp16(unsigned short* lds_p,
                                            const unsigned short* g) {
#if USE_ASYNC
  unsigned lo = (unsigned)(size_t)lds_p;      // LDS aperture: low 32 bits = offset
  asm volatile("global_load_async_to_lds_b128 %0, %1, off"
               :: "v"(lo), "v"(g) : "memory");
#else
  *(uint4*)lds_p = *(const uint4*)g;
#endif
}
static __device__ __forceinline__ void cp_drain() {
#if USE_ASYNC
  asm volatile("s_wait_asynccnt 0x0" ::: "memory");
#endif
}

// ---------------------------------------------------------------------------
// fp32 -> bf16 bulk conversion (weights).  n4 = elements/4.
// ---------------------------------------------------------------------------
__global__ void __launch_bounds__(256)
cvt_kernel(const float* __restrict__ in, unsigned short* __restrict__ out, int n4)
{
  int i = blockIdx.x * 256 + threadIdx.x;
  if (i < n4) {
    float4 f = ((const float4*)in)[i];
    uint2 p;
    p.x = pack2(f.x, f.y);
    p.y = pack2(f.z, f.w);
    ((uint2*)out)[i] = p;
  }
}

// ---------------------------------------------------------------------------
// LayerNorm over last dim (1024), fp32 in -> bf16 out. One block per token.
// ---------------------------------------------------------------------------
__global__ void __launch_bounds__(256)
ln_kernel(const float* __restrict__ x, const float* __restrict__ g,
          const float* __restrict__ bb, unsigned short* __restrict__ y)
{
  const int row  = blockIdx.x;
  const int tid  = threadIdx.x;
  const int lane = tid & 31;
  const int wv   = tid >> 5;
  const float* xr = x + (size_t)row * 1024;

  float4 v = ((const float4*)xr)[tid];
  float s = v.x + v.y + v.z + v.w;

  __shared__ float red[8];
#pragma unroll
  for (int off = 16; off > 0; off >>= 1) s += __shfl_xor(s, off);
  if (lane == 0) red[wv] = s;
  __syncthreads();
  float mu = 0.f;
#pragma unroll
  for (int i = 0; i < 8; ++i) mu += red[i];
  mu *= (1.0f / 1024.0f);
  __syncthreads();

  float d0 = v.x - mu, d1 = v.y - mu, d2 = v.z - mu, d3 = v.w - mu;
  float s2 = d0 * d0 + d1 * d1 + d2 * d2 + d3 * d3;
#pragma unroll
  for (int off = 16; off > 0; off >>= 1) s2 += __shfl_xor(s2, off);
  if (lane == 0) red[wv] = s2;
  __syncthreads();
  float var = 0.f;
#pragma unroll
  for (int i = 0; i < 8; ++i) var += red[i];
  var *= (1.0f / 1024.0f);
  const float inv = rsqrtf(var + 1e-5f);

  float4 gg = ((const float4*)g)[tid];
  float4 bbv = ((const float4*)bb)[tid];
  float y0 = d0 * inv * gg.x + bbv.x;
  float y1 = d1 * inv * gg.y + bbv.y;
  float y2 = d2 * inv * gg.z + bbv.z;
  float y3 = d3 * inv * gg.w + bbv.w;
  uint2 p; p.x = pack2(y0, y1); p.y = pack2(y2, y3);
  ((uint2*)(y + (size_t)row * 1024))[tid] = p;
}

// ---------------------------------------------------------------------------
// GEMM: C[M,N] = act(A[M,K]*B[K,N] + bias) (+ res for fp32 output).
// A,B bf16 global; double-buffered async LDS staging; output fp32 or bf16.
// Block 256 (8 waves), tile 128x128, K staged 32 at a time.
// ---------------------------------------------------------------------------
#define KP 40    // A LDS k-stride (bf16): 80B rows, 16B-aligned
#define NP 136   // B LDS n-stride (bf16): 272B rows, 16B-aligned

template <int ACT, int OUTBF>
__global__ void __launch_bounds__(256)
gemm_kernel(const unsigned short* __restrict__ A, const unsigned short* __restrict__ B,
            const float* __restrict__ bias, const float* __restrict__ res,
            float* __restrict__ Cf, unsigned short* __restrict__ Ch,
            int M, int N, int K)
{
  __shared__ __align__(16) unsigned short As[2][128 * KP];  // [m][k]
  __shared__ __align__(16) unsigned short Bs[2][32 * NP];   // [k][n] natural

  const int tid  = threadIdx.x;
  const int lane = tid & 31;
  const int wv   = tid >> 5;
  const int wm   = wv >> 1;          // 0..3
  const int wn   = wv & 1;           // 0..1
  const int hlf  = lane >> 4;        // 0/1
  const int l16  = lane & 15;
  const int bm   = blockIdx.y * 128;
  const int bn   = blockIdx.x * 128;

  auto stage = [&](int buf, int kt) {
#pragma unroll
    for (int i = 0; i < 2; ++i) {
      int fid = tid + i * 256;             // 512 x b128 for A
      int r   = fid >> 2;
      int c8  = fid & 3;
      cp16(&As[buf][r * KP + c8 * 8], A + (size_t)(bm + r) * K + kt + c8 * 8);
    }
#pragma unroll
    for (int i = 0; i < 2; ++i) {
      int fid = tid + i * 256;             // 512 x b128 for B
      int kr  = fid >> 4;
      int c8  = fid & 15;
      cp16(&Bs[buf][kr * NP + c8 * 8], B + (size_t)(kt + kr) * N + bn + c8 * 8);
    }
  };

  f32x8 acc[2][4];
#pragma unroll
  for (int mf = 0; mf < 2; ++mf)
#pragma unroll
    for (int nf = 0; nf < 4; ++nf)
#pragma unroll
      for (int i = 0; i < 8; ++i) acc[mf][nf][i] = 0.f;

  stage(0, 0);
  int cur = 0;
  for (int kt = 0; kt < K; kt += 32) {
    cp_drain();                 // my copies into tile `cur` complete
    __syncthreads();            // all waves' copies complete
    if (kt + 32 < K) stage(cur ^ 1, kt + 32);   // prefetch next tile (overlapped)

    const unsigned short* Ac = As[cur];
    const unsigned short* Bc = Bs[cur];
    BFrag a[2];
#pragma unroll
    for (int mf = 0; mf < 2; ++mf) {
      int m = wm * 32 + mf * 16 + l16;
#pragma unroll
      for (int r = 0; r < 8; ++r) {
        int kk = ((r >> 2) << 4) + (hlf << 3) + ((r & 3) << 1);
        a[mf].u[r] = *(const unsigned*)&Ac[m * KP + kk];
      }
    }
#pragma unroll
    for (int nf = 0; nf < 4; ++nf) {
      BFrag bfr;
      int n = wn * 64 + nf * 16 + l16;
#pragma unroll
      for (int r = 0; r < 8; ++r) {
        int kk = ((r >> 2) << 4) + (hlf << 3) + ((r & 3) << 1);
        bfr.u[r] = (unsigned)Bc[kk * NP + n] |
                   ((unsigned)Bc[(kk + 1) * NP + n] << 16);
      }
      acc[0][nf] = wmma_bf16(a[0], bfr, acc[0][nf]);
      acc[1][nf] = wmma_bf16(a[1], bfr, acc[1][nf]);
    }
    __syncthreads();            // all reads of `cur` done before it is re-staged
    cur ^= 1;
  }

  // ---- epilogue: bias -> (gelu) -> (residual) -> store ----
#pragma unroll
  for (int mf = 0; mf < 2; ++mf)
#pragma unroll
    for (int nf = 0; nf < 4; ++nf)
#pragma unroll
      for (int r = 0; r < 8; ++r) {
        int m = bm + wm * 32 + mf * 16 + r + hlf * 8;
        int n = bn + wn * 64 + nf * 16 + l16;
        float val = acc[mf][nf][r] + bias[n];
        if (ACT) val = gelu_exact(val);
        if (OUTBF) {
          Ch[(size_t)m * N + n] = f2bf(val);
        } else {
          if (res) val += res[(size_t)m * N + n];
          Cf[(size_t)m * N + n] = val;
        }
      }
}

// ---------------------------------------------------------------------------
// Flash-style attention, bf16 in/out.  grid=(B*H, S/64), block=128 (4 waves).
// Q staged once; K/V chunks double-buffered via async LDS-DMA.  V fragments
// are built from 16-bit column loads so no LDS transpose is needed.
// ---------------------------------------------------------------------------
#define SHD 72  // 144B rows: 16B-aligned

__global__ void __launch_bounds__(128)
attn_kernel(const unsigned short* __restrict__ Q, const unsigned short* __restrict__ Kt,
            const unsigned short* __restrict__ V, unsigned short* __restrict__ O)
{
  __shared__ __align__(16) unsigned short Qs[64 * SHD];
  __shared__ __align__(16) unsigned short Ks[2][64 * SHD];
  __shared__ __align__(16) unsigned short Vs[2][64 * SHD];
  __shared__ __align__(16) unsigned short Ps[4 * 16 * SHD];

  const int tid  = threadIdx.x;
  const int lane = tid & 31;
  const int wv   = tid >> 5;          // 0..3
  const int hlf  = lane >> 4;
  const int l16  = lane & 15;
  const int bh   = blockIdx.x;        // b*16 + h
  const int b    = bh >> 4;
  const int h    = bh & 15;
  const int q0   = blockIdx.y * 64;
  const size_t base = (size_t)b * 1024 * 1024 + (size_t)h * 64;

  auto stageKV = [&](int buf, int k0) {
#pragma unroll
    for (int i = 0; i < 4; ++i) {
      int fid = tid + i * 128;
      int r   = fid >> 3;
      int c8  = fid & 7;
      cp16(&Ks[buf][r * SHD + c8 * 8], Kt + base + (size_t)(k0 + r) * 1024 + c8 * 8);
      cp16(&Vs[buf][r * SHD + c8 * 8], V  + base + (size_t)(k0 + r) * 1024 + c8 * 8);
    }
  };

  // ---- stage Q tile (64x64 bf16), load Q fragments ----
#pragma unroll
  for (int i = 0; i < 4; ++i) {
    int fid = tid + i * 128;            // 512 x b128
    int r   = fid >> 3;
    int c8  = fid & 7;
    cp16(&Qs[r * SHD + c8 * 8], Q + base + (size_t)(q0 + r) * 1024 + c8 * 8);
  }
  cp_drain();
  __syncthreads();

  BFrag aq[2];
#pragma unroll
  for (int kc = 0; kc < 2; ++kc) {
    int m = wv * 16 + l16;
#pragma unroll
    for (int r = 0; r < 8; ++r) {
      int kk = kc * 32 + ((r >> 2) << 4) + (hlf << 3) + ((r & 3) << 1);
      aq[kc].u[r] = *(const unsigned*)&Qs[m * SHD + kk];
    }
  }

  float mrun[8], lrun[8];
  f32x8 ofrag[4];
#pragma unroll
  for (int r = 0; r < 8; ++r) { mrun[r] = -3.0e38f; lrun[r] = 0.f; }
#pragma unroll
  for (int f = 0; f < 4; ++f)
#pragma unroll
    for (int i = 0; i < 8; ++i) ofrag[f][i] = 0.f;

  stageKV(0, 0);
  int cur = 0;
  for (int c = 0; c < 16; ++c) {
    cp_drain();
    __syncthreads();
    if (c < 15) stageKV(cur ^ 1, (c + 1) * 64);    // prefetch next chunk

    const unsigned short* Kc = Ks[cur];
    const unsigned short* Vc = Vs[cur];

    // ---- S = Q K^T ----
    f32x8 s[4];
#pragma unroll
    for (int nf = 0; nf < 4; ++nf) {
#pragma unroll
      for (int i = 0; i < 8; ++i) s[nf][i] = 0.f;
#pragma unroll
      for (int kc = 0; kc < 2; ++kc) {
        BFrag bk;
        int key = nf * 16 + l16;
#pragma unroll
        for (int r = 0; r < 8; ++r) {
          int kk = kc * 32 + ((r >> 2) << 4) + (hlf << 3) + ((r & 3) << 1);
          bk.u[r] = *(const unsigned*)&Kc[key * SHD + kk];
        }
        s[nf] = wmma_bf16(aq[kc], bk, s[nf]);
      }
    }

    // ---- online softmax (row = r + 8*hlf; cols live in a 16-lane group) ----
#pragma unroll
    for (int r = 0; r < 8; ++r) {
      float mx = -3.0e38f;
#pragma unroll
      for (int nf = 0; nf < 4; ++nf) {
        s[nf][r] *= 0.125f;
        mx = fmaxf(mx, s[nf][r]);
      }
#pragma unroll
      for (int off = 1; off < 16; off <<= 1) mx = fmaxf(mx, __shfl_xor(mx, off));
      float mnew  = fmaxf(mrun[r], mx);
      float alpha = __expf(mrun[r] - mnew);
      mrun[r] = mnew;
      float rs = 0.f;
#pragma unroll
      for (int nf = 0; nf < 4; ++nf) {
        float p = __expf(s[nf][r] - mnew);
        s[nf][r] = p;
        rs += p;
      }
#pragma unroll
      for (int off = 1; off < 16; off <<= 1) rs += __shfl_xor(rs, off);
      lrun[r] = lrun[r] * alpha + rs;
#pragma unroll
      for (int f = 0; f < 4; ++f) ofrag[f][r] *= alpha;
    }

    // ---- spill P (C layout) -> per-wave LDS -> reload in A layout ----
    unsigned short* pw = &Ps[wv * 16 * SHD];
#pragma unroll
    for (int nf = 0; nf < 4; ++nf)
#pragma unroll
      for (int r = 0; r < 8; ++r)
        pw[(r + hlf * 8) * SHD + nf * 16 + l16] = f2bf(s[nf][r]);

    BFrag pa[2];
#pragma unroll
    for (int kc = 0; kc < 2; ++kc)
#pragma unroll
      for (int r = 0; r < 8; ++r) {
        int kk = kc * 32 + ((r >> 2) << 4) + (hlf << 3) + ((r & 3) << 1);
        pa[kc].u[r] = *(const unsigned*)&pw[l16 * SHD + kk];
      }

    // ---- O += P V   (V fragments: 16-bit column loads, no transpose) ----
#pragma unroll
    for (int hf = 0; hf < 4; ++hf)
#pragma unroll
      for (int kc = 0; kc < 2; ++kc) {
        BFrag bvf;
        int hd = hf * 16 + l16;
#pragma unroll
        for (int r = 0; r < 8; ++r) {
          int key = kc * 32 + ((r >> 2) << 4) + (hlf << 3) + ((r & 3) << 1);
          bvf.u[r] = (unsigned)Vc[key * SHD + hd] |
                     ((unsigned)Vc[(key + 1) * SHD + hd] << 16);
        }
        ofrag[hf] = wmma_bf16(pa[kc], bvf, ofrag[hf]);
      }

    __syncthreads();            // all reads of `cur` done before re-staging
    cur ^= 1;
  }

  // ---- normalize, write bf16 out ----
#pragma unroll
  for (int hf = 0; hf < 4; ++hf)
#pragma unroll
    for (int r = 0; r < 8; ++r) {
      int m = q0 + wv * 16 + r + hlf * 8;
      int n = h * 64 + hf * 16 + l16;
      O[(size_t)b * 1024 * 1024 + (size_t)m * 1024 + n] = f2bf(ofrag[hf][r] / lrun[r]);
    }
}

// ---------------------------------------------------------------------------
// Host-side orchestration.
// ---------------------------------------------------------------------------
extern "C" void kernel_launch(void* const* d_in, const int* in_sizes, int n_in,
                              void* d_out, int out_size, void* d_ws, size_t ws_size,
                              hipStream_t stream)
{
  (void)in_sizes; (void)n_in; (void)out_size; (void)ws_size;
  const float* x   = (const float*)d_in[0];
  const float* wq  = (const float*)d_in[1];  const float* bq  = (const float*)d_in[2];
  const float* wk  = (const float*)d_in[3];  const float* bk  = (const float*)d_in[4];
  const float* wvw = (const float*)d_in[5];  const float* bv  = (const float*)d_in[6];
  const float* wo  = (const float*)d_in[7];  const float* bo  = (const float*)d_in[8];
  const float* w1  = (const float*)d_in[9];  const float* b1  = (const float*)d_in[10];
  const float* w2  = (const float*)d_in[11]; const float* b2  = (const float*)d_in[12];
  const float* g1  = (const float*)d_in[13]; const float* lb1 = (const float*)d_in[14];
  const float* g2  = (const float*)d_in[15]; const float* lb2 = (const float*)d_in[16];
  float* out = (float*)d_out;

  // workspace layout (ushort units), total 36M ushort = 72 MB:
  //  wqb wkb wvb wob : [0,4M)   w1b:[4M,8M)  w2b:[8M,12M)
  //  h1/attn:[12M,16M)  q/h2:[16M,20M)  k:[20M,24M)  v:[24M,28M)
  //  f1:[20M,36M)  (reuses k,v after attention)
  unsigned short* wsb = (unsigned short*)d_ws;
  const size_t M1 = (size_t)1 << 20;
  unsigned short* wqb = wsb;
  unsigned short* wkb = wsb + 1 * M1;
  unsigned short* wvb = wsb + 2 * M1;
  unsigned short* wob = wsb + 3 * M1;
  unsigned short* w1b = wsb + 4 * M1;
  unsigned short* w2b = wsb + 8 * M1;
  unsigned short* h1  = wsb + 12 * M1;
  unsigned short* qb  = wsb + 16 * M1;
  unsigned short* kb  = wsb + 20 * M1;
  unsigned short* vb  = wsb + 24 * M1;
  unsigned short* f1  = wsb + 20 * M1;

  const dim3 blk256(256);
  const dim3 gD(1024 / 128, 4096 / 128);
  const dim3 gF(4096 / 128, 4096 / 128);

  // 0) weights -> bf16 (once per launch; bandwidth-bound)
  cvt_kernel<<<1024, blk256, 0, stream>>>(wq,  wqb, 1 << 18);
  cvt_kernel<<<1024, blk256, 0, stream>>>(wk,  wkb, 1 << 18);
  cvt_kernel<<<1024, blk256, 0, stream>>>(wvw, wvb, 1 << 18);
  cvt_kernel<<<1024, blk256, 0, stream>>>(wo,  wob, 1 << 18);
  cvt_kernel<<<4096, blk256, 0, stream>>>(w1,  w1b, 1 << 20);
  cvt_kernel<<<4096, blk256, 0, stream>>>(w2,  w2b, 1 << 20);

  // 1) h1 = LN1(x)
  ln_kernel<<<4096, blk256, 0, stream>>>(x, g1, lb1, h1);
  // 2) Q,K,V projections (bf16 out)
  gemm_kernel<0, 1><<<gD, blk256, 0, stream>>>(h1, wqb, bq, nullptr, nullptr, qb, 4096, 1024, 1024);
  gemm_kernel<0, 1><<<gD, blk256, 0, stream>>>(h1, wkb, bk, nullptr, nullptr, kb, 4096, 1024, 1024);
  gemm_kernel<0, 1><<<gD, blk256, 0, stream>>>(h1, wvb, bv, nullptr, nullptr, vb, 4096, 1024, 1024);
  // 3) attention -> h1 (bf16)
  attn_kernel<<<dim3(64, 16), dim3(128), 0, stream>>>(qb, kb, vb, h1);
  // 4) out = x + attn @ wo + bo   (fp32 residual stream)
  gemm_kernel<0, 0><<<gD, blk256, 0, stream>>>(h1, wob, bo, x, out, nullptr, 4096, 1024, 1024);
  // 5) h2 = LN2(out) -> qb
  ln_kernel<<<4096, blk256, 0, stream>>>(out, g2, lb2, qb);
  // 6) f1 = gelu(h2 @ w1 + b1)  (bf16 out)
  gemm_kernel<1, 1><<<gF, blk256, 0, stream>>>(qb, w1b, b1, nullptr, nullptr, f1, 4096, 4096, 1024);
  // 7) out = out + f1 @ w2 + b2
  gemm_kernel<0, 0><<<gD, blk256, 0, stream>>>(f1, w2b, b2, out, out, nullptr, 4096, 1024, 4096);
}
